// myGLMixBlock_26087631356175
// MI455X (gfx1250) — matrix operand order repaired and checked by reading, not compiled, not run on hardware
//
#include <hip/hip_runtime.h>
#include <math.h>

// ---------------------------------------------------------------------------
// Problem constants (from reference setup_inputs)
// ---------------------------------------------------------------------------
#define B_      16
#define C_      256
#define H_      64
#define W_      64
#define N_      4096      // H_*W_
#define S_      64        // 8x8 slots
#define NH_     8         // heads
#define DH_     32        // head dim
#define HID_    512       // FRFN hidden
#define THID_   1024      // 2*HID

typedef _Float16 h16;
typedef __attribute__((ext_vector_type(8)))  _Float16 v8h;
typedef __attribute__((ext_vector_type(16))) _Float16 v16h;
typedef __attribute__((ext_vector_type(8)))  float    v8f;

// ---------------------------------------------------------------------------
// WMMA helpers (v_wmma_f32_16x16x32_f16, wave32, ISA 7.12.2 layouts)
// ---------------------------------------------------------------------------
__device__ __forceinline__ v8f wmma32(v16h a, v16h b, v8f c) {
  return __builtin_amdgcn_wmma_f32_16x16x32_f16(
      /*neg_a=*/false, a, /*neg_b=*/false, b,
      /*c_mod=*/(short)0, c, /*reuse_a=*/false, /*reuse_b=*/false);
}

// A tile: row-major, K contiguous.  lane L: row m=L&15, half=L>>4.
__device__ __forceinline__ v16h frag_a(const h16* A, int lda, int kBase, int lane) {
  const int m = lane & 15, hf = lane >> 4;
  const h16* p = A + (size_t)m * lda + kBase + hf * 8;
  v8h lo = *(const v8h*)p;
  v8h hi = *(const v8h*)(p + 16);
  v16h r;
#pragma unroll
  for (int i = 0; i < 8; ++i) { r[i] = lo[i]; r[8 + i] = hi[i]; }
  return r;
}

// B transposed (row-major N x K, K contiguous): element(k,n) = Bt[n*ldb+k]
__device__ __forceinline__ v16h frag_bt(const h16* Bt, int ldb, int kBase, int lane) {
  const int n = lane & 15, hf = lane >> 4;
  const h16* p = Bt + (size_t)n * ldb + kBase + hf * 8;
  v8h lo = *(const v8h*)p;
  v8h hi = *(const v8h*)(p + 16);
  v16h r;
#pragma unroll
  for (int i = 0; i < 8; ++i) { r[i] = lo[i]; r[8 + i] = hi[i]; }
  return r;
}

__device__ __forceinline__ float gelu_f(float v) {
  return 0.5f * v * (1.0f + erff(v * 0.7071067811865476f));
}

// ---------------------------------------------------------------------------
// Utility kernels
// ---------------------------------------------------------------------------
__global__ void k_f2h(const float* __restrict__ s, h16* __restrict__ d, int n) {
  int i = blockIdx.x * blockDim.x + threadIdx.x;
  if (i < n) d[i] = (h16)s[i];
}

// pooled 8x8 slot means + L2 normalize over C.  One block per (b,s), 256 thr.
__global__ void k_pool_slotsn(const float* __restrict__ x, h16* __restrict__ slotsn) {
  int bs = blockIdx.x;
  int c  = threadIdx.x;
  int b = bs / S_, s = bs % S_;
  int sy = s >> 3, sx = s & 7;
  const float* p = x + ((size_t)b * C_ + c) * N_;
  float acc = 0.f;
#pragma unroll 4
  for (int py = 0; py < 8; ++py)
    for (int px = 0; px < 8; ++px)
      acc += p[(sy * 8 + py) * W_ + sx * 8 + px];
  acc *= (1.0f / 64.0f);
  __shared__ float red[256];
  red[c] = acc * acc;
  __syncthreads();
  for (int off = 128; off > 0; off >>= 1) {
    if (c < off) red[c] += red[c + off];
    __syncthreads();
  }
  float rn = 1.0f / fmaxf(sqrtf(red[0]), 1e-12f);
  slotsn[(size_t)bs * C_ + c] = (h16)(acc * rn);
}

// LayerNorm over C (NCHW input) -> xf (NHWC f16) and xfn = l2norm(xf)
__global__ void k_ln1(const float* __restrict__ x, const float* __restrict__ w1,
                      const float* __restrict__ b1, h16* __restrict__ xf,
                      h16* __restrict__ xfn) {
  size_t idx = (size_t)blockIdx.x * blockDim.x + threadIdx.x;  // b*N + n
  if (idx >= (size_t)B_ * N_) return;
  size_t b = idx / N_, n = idx % N_;
  const float* p = x + b * (size_t)C_ * N_ + n;
  float s = 0.f, s2 = 0.f;
  for (int c = 0; c < C_; ++c) { float v = p[(size_t)c * N_]; s += v; s2 += v * v; }
  float mu = s * (1.0f / C_);
  float var = s2 * (1.0f / C_) - mu * mu;
  float rs = rsqrtf(var + 1e-5f);
  h16* q = xf + idx * C_;
  float nrm = 0.f;
  for (int c = 0; c < C_; ++c) {
    float v = (p[(size_t)c * N_] - mu) * rs * w1[c] + b1[c];
    q[c] = (h16)v;
    nrm += v * v;
  }
  float rn = 1.0f / fmaxf(sqrtf(nrm), 1e-12f);
  h16* qn = xfn + idx * C_;
  for (int c = 0; c < C_; ++c) qn[c] = (h16)((float)q[c] * rn);
}

// LayerNorm over C (NCHW f32 input) -> NHWC f16
__global__ void k_ln2(const float* __restrict__ x, const float* __restrict__ w2,
                      const float* __restrict__ b2, h16* __restrict__ xln) {
  size_t idx = (size_t)blockIdx.x * blockDim.x + threadIdx.x;
  if (idx >= (size_t)B_ * N_) return;
  size_t b = idx / N_, n = idx % N_;
  const float* p = x + b * (size_t)C_ * N_ + n;
  float s = 0.f, s2 = 0.f;
  for (int c = 0; c < C_; ++c) { float v = p[(size_t)c * N_]; s += v; s2 += v * v; }
  float mu = s * (1.0f / C_);
  float var = s2 * (1.0f / C_) - mu * mu;
  float rs = rsqrtf(var + 1e-5f);
  h16* q = xln + idx * C_;
  for (int c = 0; c < C_; ++c)
    q[c] = (h16)((p[(size_t)c * N_] - mu) * rs * w2[c] + b2[c]);
}

// 16x16 LDS-tiled transpose: xf (b,n,c) f16 -> xfT (b,c,n) f16
__global__ void k_transp_xf(const h16* __restrict__ xf, h16* __restrict__ xfT) {
  __shared__ h16 tile[16][17];
  int bt = blockIdx.x;                 // b*(N/16)*(C/16) blocks
  int cb = bt & (C_ / 16 - 1);
  int nb = (bt >> 4) & (N_ / 16 - 1);
  int b  = bt >> 12;
  int t = threadIdx.x;
  int i = t >> 4, j = t & 15;          // load: row n-local i, col c-local j
  tile[i][j] = xf[((size_t)b * N_ + nb * 16 + i) * C_ + cb * 16 + j];
  __syncthreads();
  int j2 = t >> 4, i2 = t & 15;        // store: row c-local j2, col n-local i2
  xfT[(size_t)b * C_ * N_ + (size_t)(cb * 16 + j2) * N_ + nb * 16 + i2] = tile[i2][j2];
}

// vT[(b*8+h), d, k] = qkv[b, k, 2C + h*32 + d]
__global__ void k_transp_v(const h16* __restrict__ qkv, h16* __restrict__ vT) {
  int idx = blockIdx.x * blockDim.x + threadIdx.x;   // total B*NH*DH*S
  if (idx >= B_ * NH_ * DH_ * S_) return;
  int k = idx & 63;
  int d = (idx >> 6) & 31;
  int h = (idx >> 11) & 7;
  int b = idx >> 14;
  vT[idx] = qkv[((size_t)b * S_ + k) * (3 * C_) + 2 * C_ + h * DH_ + d];
}

// softmax over n=4096 per (b,s) row; one block per row
__global__ void k_softmax_n(const float* __restrict__ logits, h16* __restrict__ attn) {
  int bs = blockIdx.x, t = threadIdx.x;
  const float* row = logits + (size_t)bs * N_;
  __shared__ float red[256];
  float m = -1e30f;
  for (int i = t; i < N_; i += 256) m = fmaxf(m, row[i]);
  red[t] = m; __syncthreads();
  for (int off = 128; off > 0; off >>= 1) {
    if (t < off) red[t] = fmaxf(red[t], red[t + off]);
    __syncthreads();
  }
  m = red[0]; __syncthreads();
  float s = 0.f;
  for (int i = t; i < N_; i += 256) s += expf(row[i] - m);
  red[t] = s; __syncthreads();
  for (int off = 128; off > 0; off >>= 1) {
    if (t < off) red[t] += red[t + off];
    __syncthreads();
  }
  float inv = 1.0f / red[0];
  h16* o = attn + (size_t)bs * N_;
  for (int i = t; i < N_; i += 256) o[i] = (h16)(expf(row[i] - m) * inv);
}

// softmax over s=64 of logits^T per (b,n); one thread per row
__global__ void k_softmax_T(const float* __restrict__ logits, h16* __restrict__ attnT) {
  size_t idx = (size_t)blockIdx.x * blockDim.x + threadIdx.x;
  if (idx >= (size_t)B_ * N_) return;
  size_t b = idx / N_, n = idx % N_;
  const float* p = logits + b * (size_t)S_ * N_ + n;
  float m = -1e30f;
  for (int s = 0; s < S_; ++s) m = fmaxf(m, p[(size_t)s * N_]);
  float sum = 0.f;
  for (int s = 0; s < S_; ++s) sum += expf(p[(size_t)s * N_] - m);
  float inv = 1.0f / sum;
  h16* o = attnT + idx * S_;
  for (int s = 0; s < S_; ++s) o[s] = (h16)(expf(p[(size_t)s * N_] - m) * inv);
}

// per-head attention softmax (scale 1/sqrt(32)); one thread per 64-wide row
__global__ void k_softmax_h(const float* __restrict__ scores, h16* __restrict__ probs) {
  int idx = blockIdx.x * blockDim.x + threadIdx.x;
  if (idx >= B_ * NH_ * S_) return;
  const float* p = scores + (size_t)idx * S_;
  const float sc = 0.17677669529663687f;
  float m = -1e30f;
  for (int j = 0; j < S_; ++j) m = fmaxf(m, p[j] * sc);
  float s = 0.f;
  for (int j = 0; j < S_; ++j) s += expf(p[j] * sc - m);
  float inv = 1.0f / s;
  h16* o = probs + (size_t)idx * S_;
  for (int j = 0; j < S_; ++j) o[j] = (h16)(expf(p[j] * sc - m) * inv);
}

// partial 3x3 conv over first 64 channels of xln (NHWC f16), copy rest
__global__ void k_pconv(const h16* __restrict__ xln, const float* __restrict__ pw,
                        h16* __restrict__ xcat) {
  size_t idx = (size_t)blockIdx.x * blockDim.x + threadIdx.x;
  if (idx >= (size_t)B_ * N_ * C_) return;
  int c = (int)(idx & (C_ - 1));
  size_t bn = idx >> 8;
  int n = (int)(bn % N_);
  int b = (int)(bn / N_);
  if (c >= 64) { xcat[idx] = xln[idx]; return; }
  int hy = n >> 6, wx = n & 63;
  float acc = 0.f;
  for (int ic = 0; ic < 64; ++ic) {
    const float* wp = pw + ((size_t)c * 64 + ic) * 9;
#pragma unroll
    for (int ky = 0; ky < 3; ++ky) {
      int y = hy + ky - 1;
      if ((unsigned)y >= (unsigned)H_) continue;
#pragma unroll
      for (int kx = 0; kx < 3; ++kx) {
        int xx = wx + kx - 1;
        if ((unsigned)xx >= (unsigned)W_) continue;
        acc += (float)xln[((size_t)b * N_ + y * W_ + xx) * C_ + ic] * wp[ky * 3 + kx];
      }
    }
  }
  xcat[idx] = (h16)acc;
}

// depthwise 3x3 conv on y1 + bias + gelu, gated by y2 -> p
__global__ void k_dwconv(const h16* __restrict__ y1, const h16* __restrict__ y2,
                         const float* __restrict__ dww, const float* __restrict__ dwb,
                         h16* __restrict__ p) {
  size_t idx = (size_t)blockIdx.x * blockDim.x + threadIdx.x;
  if (idx >= (size_t)B_ * N_ * HID_) return;
  int ch = (int)(idx & (HID_ - 1));
  size_t bn = idx >> 9;
  int n = (int)(bn % N_);
  int b = (int)(bn / N_);
  int hy = n >> 6, wx = n & 63;
  float acc = dwb[ch];
  const float* wp = dww + (size_t)ch * 9;
#pragma unroll
  for (int ky = 0; ky < 3; ++ky) {
    int y = hy + ky - 1;
    if ((unsigned)y >= (unsigned)H_) continue;
#pragma unroll
    for (int kx = 0; kx < 3; ++kx) {
      int xx = wx + kx - 1;
      if ((unsigned)xx >= (unsigned)W_) continue;
      acc += (float)y1[((size_t)b * N_ + y * W_ + xx) * HID_ + ch] * wp[ky * 3 + kx];
    }
  }
  p[idx] = (h16)(gelu_f(acc) * (float)y2[idx]);
}

// ---------------------------------------------------------------------------
// WMMA GEMM kernels.  Wide kernels: one 16x64 strip per wave (A-frag reused
// across 4 WMMAs); 8 waves / 256-thread block.
// ---------------------------------------------------------------------------

// logits = scale * slotsn(64x256) @ xfn^T -> (b,64,4096) f32
__global__ void k_gemm_logits(const h16* __restrict__ An, const h16* __restrict__ Btn,
                              const float* __restrict__ scalep, float* __restrict__ out) {
  int lane = threadIdx.x & 31, wave = threadIdx.x >> 5;
  int g = blockIdx.x * 8 + wave;
  const int gN = N_ / 64, tilesM = S_ / 16, perB = tilesM * gN;   // 64, 4, 256
  if (g >= B_ * perB) return;
  int b = g / perB, t = g % perB, tm = t / gN, tg = t % gN;
  const h16* A  = An  + (size_t)b * S_ * C_ + (size_t)(tm * 16) * C_;
  const h16* Bt = Btn + (size_t)b * N_ * C_ + (size_t)(tg * 64) * C_;
  v8f acc[4] = {{}, {}, {}, {}};
#pragma unroll
  for (int k = 0; k < C_; k += 32) {
    v16h af = frag_a(A, C_, k, lane);
#pragma unroll
    for (int q = 0; q < 4; ++q)
      acc[q] = wmma32(af, frag_bt(Bt + (size_t)(q * 16) * C_, C_, k, lane), acc[q]);
  }
  float scale = scalep[0];
  int col = lane & 15, hf = lane >> 4;
#pragma unroll
  for (int q = 0; q < 4; ++q) {
    float* O = out + (size_t)b * S_ * N_ + (size_t)(tm * 16) * N_ + tg * 64 + q * 16;
#pragma unroll
    for (int r = 0; r < 8; ++r) O[(size_t)(r + 8 * hf) * N_ + col] = acc[q][r] * scale;
  }
}

// slots = attn(64x4096) @ xf(4096x256), B supplied transposed as xfT (b,c,n)
__global__ void k_gemm_slots(const h16* __restrict__ attn, const h16* __restrict__ xfT,
                             h16* __restrict__ slots) {
  int lane = threadIdx.x & 31, wave = threadIdx.x >> 5;
  int g = blockIdx.x * 8 + wave;
  const int gN = C_ / 64, tilesM = S_ / 16, perB = tilesM * gN;   // 4, 4, 16
  if (g >= B_ * perB) return;
  int b = g / perB, t = g % perB, tm = t / gN, tg = t % gN;
  const h16* A  = attn + (size_t)b * S_ * N_ + (size_t)(tm * 16) * N_;
  const h16* Bt = xfT  + (size_t)b * C_ * N_ + (size_t)(tg * 64) * N_;
  v8f acc[4] = {{}, {}, {}, {}};
  for (int k = 0; k < N_; k += 32) {
    __builtin_prefetch((const void*)(A + k + 256), 0, 1);
    v16h af = frag_a(A, N_, k, lane);
#pragma unroll
    for (int q = 0; q < 4; ++q)
      acc[q] = wmma32(af, frag_bt(Bt + (size_t)(q * 16) * N_, N_, k, lane), acc[q]);
  }
  int col = lane & 15, hf = lane >> 4;
#pragma unroll
  for (int q = 0; q < 4; ++q) {
    h16* O = slots + (size_t)b * S_ * C_ + (size_t)(tm * 16) * C_ + tg * 64 + q * 16;
#pragma unroll
    for (int r = 0; r < 8; ++r) O[(size_t)(r + 8 * hf) * C_ + col] = (h16)acc[q][r];
  }
}

// qkv = slots(64x256) @ Wi^T + bi -> (b,64,768) f16
__global__ void k_gemm_qkv(const h16* __restrict__ slots, const h16* __restrict__ wi,
                           const float* __restrict__ bi, h16* __restrict__ qkv) {
  int lane = threadIdx.x & 31, wave = threadIdx.x >> 5;
  int g = blockIdx.x * 8 + wave;
  const int gN = (3 * C_) / 64, tilesM = S_ / 16, perB = tilesM * gN;  // 12, 4, 48
  if (g >= B_ * perB) return;
  int b = g / perB, t = g % perB, tm = t / gN, tg = t % gN;
  const h16* A  = slots + (size_t)b * S_ * C_ + (size_t)(tm * 16) * C_;
  const h16* Bt = wi + (size_t)(tg * 64) * C_;
  v8f acc[4] = {{}, {}, {}, {}};
#pragma unroll
  for (int k = 0; k < C_; k += 32) {
    v16h af = frag_a(A, C_, k, lane);
#pragma unroll
    for (int q = 0; q < 4; ++q)
      acc[q] = wmma32(af, frag_bt(Bt + (size_t)(q * 16) * C_, C_, k, lane), acc[q]);
  }
  int col = lane & 15, hf = lane >> 4;
#pragma unroll
  for (int q = 0; q < 4; ++q) {
    int o = tg * 64 + q * 16 + col;
    float bias = bi[o];
    h16* O = qkv + (size_t)b * S_ * (3 * C_) + (size_t)(tm * 16) * (3 * C_) + o;
#pragma unroll
    for (int r = 0; r < 8; ++r)
      O[(size_t)(r + 8 * hf) * (3 * C_)] = (h16)(acc[q][r] + bias);
  }
}

// per-head scores = q(64x32) @ k^T -> (b,h,64,64) f32
__global__ void k_gemm_scores(const h16* __restrict__ qkv, float* __restrict__ scores) {
  int lane = threadIdx.x & 31, wave = threadIdx.x >> 5;
  int tile = blockIdx.x * 8 + wave;               // B*NH*16
  if (tile >= B_ * NH_ * 16) return;
  int bh = tile >> 4, tt = tile & 15, tm = tt >> 2, tn = tt & 3;
  int b = bh >> 3, h = bh & 7;
  const h16* A  = qkv + (size_t)b * S_ * (3 * C_) + (size_t)(tm * 16) * (3 * C_) + h * DH_;
  const h16* Bt = qkv + (size_t)b * S_ * (3 * C_) + (size_t)(tn * 16) * (3 * C_) + C_ + h * DH_;
  v8f acc = {};
  acc = wmma32(frag_a(A, 3 * C_, 0, lane), frag_bt(Bt, 3 * C_, 0, lane), acc);
  int col = lane & 15, hf = lane >> 4;
  float* O = scores + ((size_t)bh * S_ + tm * 16) * S_ + tn * 16;
#pragma unroll
  for (int r = 0; r < 8; ++r) O[(size_t)(r + 8 * hf) * S_ + col] = acc[r];
}

// o = probs(64x64) @ v(64x32), V supplied transposed as vT (bh,32,64)
__global__ void k_gemm_av(const h16* __restrict__ probs, const h16* __restrict__ vT,
                          h16* __restrict__ oh) {
  int lane = threadIdx.x & 31, wave = threadIdx.x >> 5;
  int tile = blockIdx.x * 8 + wave;               // B*NH*8
  if (tile >= B_ * NH_ * 8) return;
  int bh = tile >> 3, tt = tile & 7, tm = tt >> 1, tn = tt & 1;
  int b = bh >> 3, h = bh & 7;
  const h16* A  = probs + (size_t)bh * S_ * S_ + (size_t)(tm * 16) * S_;
  const h16* Bt = vT + (size_t)bh * DH_ * S_ + (size_t)(tn * 16) * S_;
  v8f acc = {};
#pragma unroll
  for (int k = 0; k < S_; k += 32)
    acc = wmma32(frag_a(A, S_, k, lane), frag_bt(Bt, S_, k, lane), acc);
  int col = lane & 15, hf = lane >> 4;
  h16* O = oh + (size_t)b * S_ * C_ + (size_t)(tm * 16) * C_ + h * DH_ + tn * 16;
#pragma unroll
  for (int r = 0; r < 8; ++r) O[(size_t)(r + 8 * hf) * C_ + col] = (h16)acc[r];
}

// slotsmT = (o(64x256) @ Wo^T + bo)^T -> (b,256,64) f16 (transposed epilogue)
__global__ void k_gemm_oproj(const h16* __restrict__ oh, const h16* __restrict__ wo,
                             const float* __restrict__ bo, h16* __restrict__ slotsmT) {
  int lane = threadIdx.x & 31, wave = threadIdx.x >> 5;
  int g = blockIdx.x * 8 + wave;
  const int gN = C_ / 64, tilesM = S_ / 16, perB = tilesM * gN;   // 4, 4, 16
  if (g >= B_ * perB) return;
  int b = g / perB, t = g % perB, tm = t / gN, tg = t % gN;
  const h16* A  = oh + (size_t)b * S_ * C_ + (size_t)(tm * 16) * C_;
  const h16* Bt = wo + (size_t)(tg * 64) * C_;
  v8f acc[4] = {{}, {}, {}, {}};
#pragma unroll
  for (int k = 0; k < C_; k += 32) {
    v16h af = frag_a(A, C_, k, lane);
#pragma unroll
    for (int q = 0; q < 4; ++q)
      acc[q] = wmma32(af, frag_bt(Bt + (size_t)(q * 16) * C_, C_, k, lane), acc[q]);
  }
  int col = lane & 15, hf = lane >> 4;
#pragma unroll
  for (int q = 0; q < 4; ++q) {
    int c = tg * 64 + q * 16 + col;
    float bias = bo[c];
#pragma unroll
    for (int r = 0; r < 8; ++r) {
      int s = tm * 16 + r + 8 * hf;
      slotsmT[(size_t)b * C_ * S_ + (size_t)c * S_ + s] = (h16)(acc[q][r] + bias);
    }
  }
}

// out = attnT(4096x64) @ slotsm, B transposed (b,256,64); x1 = x + out (NCHW)
__global__ void k_gemm_out(const h16* __restrict__ attnT, const h16* __restrict__ slotsmT,
                           const float* __restrict__ x0, float* __restrict__ x1) {
  int lane = threadIdx.x & 31, wave = threadIdx.x >> 5;
  int g = blockIdx.x * 8 + wave;
  const int gN = C_ / 64, tilesM = N_ / 16, perB = tilesM * gN;   // 4, 256, 1024
  if (g >= B_ * perB) return;
  int b = g / perB, t = g % perB, tm = t / gN, tg = t % gN;
  const h16* A  = attnT   + (size_t)b * N_ * S_ + (size_t)(tm * 16) * S_;
  const h16* Bt = slotsmT + (size_t)b * C_ * S_ + (size_t)(tg * 64) * S_;
  v8f acc[4] = {{}, {}, {}, {}};
#pragma unroll
  for (int k = 0; k < S_; k += 32) {
    v16h af = frag_a(A, S_, k, lane);
#pragma unroll
    for (int q = 0; q < 4; ++q)
      acc[q] = wmma32(af, frag_bt(Bt + (size_t)(q * 16) * S_, S_, k, lane), acc[q]);
  }
  int col = lane & 15, hf = lane >> 4;
#pragma unroll
  for (int q = 0; q < 4; ++q) {
    int c = tg * 64 + q * 16 + col;
#pragma unroll
    for (int r = 0; r < 8; ++r) {
      int n = tm * 16 + r + 8 * hf;
      size_t oi = (size_t)b * C_ * N_ + (size_t)c * N_ + n;
      x1[oi] = x0[oi] + acc[q][r];
    }
  }
}

// y = gelu(xcat(4096x256) @ lin1_w^T + b); split into y1|y2
__global__ void k_gemm_lin1(const h16* __restrict__ xcat, const h16* __restrict__ l1w,
                            const float* __restrict__ l1b, h16* __restrict__ y1,
                            h16* __restrict__ y2) {
  int lane = threadIdx.x & 31, wave = threadIdx.x >> 5;
  int g = blockIdx.x * 8 + wave;
  const int gN = THID_ / 64, tilesM = N_ / 16, perB = tilesM * gN;  // 16, 256, 4096
  if (g >= B_ * perB) return;
  int b = g / perB, t = g % perB, tm = t / gN, tg = t % gN;
  const h16* A  = xcat + (size_t)b * N_ * C_ + (size_t)(tm * 16) * C_;
  const h16* Bt = l1w + (size_t)(tg * 64) * C_;
  v8f acc[4] = {{}, {}, {}, {}};
#pragma unroll
  for (int k = 0; k < C_; k += 32) {
    __builtin_prefetch((const void*)(A + k + 64), 0, 1);
    v16h af = frag_a(A, C_, k, lane);
#pragma unroll
    for (int q = 0; q < 4; ++q)
      acc[q] = wmma32(af, frag_bt(Bt + (size_t)(q * 16) * C_, C_, k, lane), acc[q]);
  }
  int col = lane & 15, hf = lane >> 4;
#pragma unroll
  for (int q = 0; q < 4; ++q) {
    int o = tg * 64 + q * 16 + col;
    float bias = l1b[o];
#pragma unroll
    for (int r = 0; r < 8; ++r) {
      int n = tm * 16 + r + 8 * hf;
      float gv = gelu_f(acc[q][r] + bias);
      size_t base = ((size_t)b * N_ + n) * HID_;
      if (o < HID_) y1[base + o] = (h16)gv;
      else          y2[base + (o - HID_)] = (h16)gv;
    }
  }
}

// z = p(4096x512) @ lin2_w^T + b + x1; out = z ; out2 = x_orig - z
__global__ void k_gemm_lin2(const h16* __restrict__ p, const h16* __restrict__ l2w,
                            const float* __restrict__ l2b, const float* __restrict__ x1,
                            const float* __restrict__ x0, float* __restrict__ out,
                            float* __restrict__ out2) {
  int lane = threadIdx.x & 31, wave = threadIdx.x >> 5;
  int g = blockIdx.x * 8 + wave;
  const int gN = C_ / 64, tilesM = N_ / 16, perB = tilesM * gN;   // 4, 256, 1024
  if (g >= B_ * perB) return;
  int b = g / perB, t = g % perB, tm = t / gN, tg = t % gN;
  const h16* A  = p + (size_t)b * N_ * HID_ + (size_t)(tm * 16) * HID_;
  const h16* Bt = l2w + (size_t)(tg * 64) * HID_;
  v8f acc[4] = {{}, {}, {}, {}};
#pragma unroll
  for (int k = 0; k < HID_; k += 32) {
    __builtin_prefetch((const void*)(A + k + 64), 0, 1);
    v16h af = frag_a(A, HID_, k, lane);
#pragma unroll
    for (int q = 0; q < 4; ++q)
      acc[q] = wmma32(af, frag_bt(Bt + (size_t)(q * 16) * HID_, HID_, k, lane), acc[q]);
  }
  int col = lane & 15, hf = lane >> 4;
#pragma unroll
  for (int q = 0; q < 4; ++q) {
    int c = tg * 64 + q * 16 + col;
    float bias = l2b[c];
#pragma unroll
    for (int r = 0; r < 8; ++r) {
      int n = tm * 16 + r + 8 * hf;
      size_t oi = (size_t)b * C_ * N_ + (size_t)c * N_ + n;
      float val = acc[q][r] + bias + x1[oi];
      out[oi]  = val;
      out2[oi] = x0[oi] - val;
    }
  }
}

// ---------------------------------------------------------------------------
// Host launcher
// ---------------------------------------------------------------------------
extern "C" void kernel_launch(void* const* d_in, const int* in_sizes, int n_in,
                              void* d_out, int out_size, void* d_ws, size_t ws_size,
                              hipStream_t stream) {
  (void)in_sizes; (void)n_in; (void)out_size; (void)ws_size;
  const float* x      = (const float*)d_in[0];
  const float* scale  = (const float*)d_in[1];
  const float* n1w    = (const float*)d_in[2];
  const float* n1b    = (const float*)d_in[3];
  const float* inpw   = (const float*)d_in[4];
  const float* inpb   = (const float*)d_in[5];
  const float* outpw  = (const float*)d_in[6];
  const float* outpb  = (const float*)d_in[7];
  const float* n2w    = (const float*)d_in[8];
  const float* n2b    = (const float*)d_in[9];
  const float* pconvw = (const float*)d_in[10];
  const float* lin1w  = (const float*)d_in[11];
  const float* lin1b  = (const float*)d_in[12];
  const float* dww    = (const float*)d_in[13];
  const float* dwb    = (const float*)d_in[14];
  const float* lin2w  = (const float*)d_in[15];
  const float* lin2b  = (const float*)d_in[16];

  float* out  = (float*)d_out;
  float* out2 = out + (size_t)B_ * C_ * N_;

  // workspace arena (phase-overlapped regions)
  char* w = (char*)d_ws;
  const size_t NCB = (size_t)B_ * N_ * C_ * 2;           // 33,554,432 B (f16 NHWC)
  // BUF_A [0, 2*NCB): xf | xfn (xfn later overwritten by xfT) ; later y1
  h16* xf    = (h16*)(w + 0);
  h16* xfn   = (h16*)(w + NCB);
  h16* xfT   = (h16*)(w + NCB);   // reuses xfn region after logits GEMM
  h16* y1    = (h16*)(w + 0);
  // BUF_B [2*NCB, 4*NCB): attention scratch ; later y2
  char* bb = w + 2 * NCB;
  float* logits   = (float*)(bb + 0);                    // 16,777,216 B
  h16*   attn     = (h16*)(bb + 16777216);               //  8,388,608 B
  h16*   attnT    = (h16*)(bb + 25165824);               //  8,388,608 B
  h16*   qkv      = (h16*)(bb + 33554432);               //  1,572,864 B
  float* scores   = (float*)(bb + 35127296);             //  2,097,152 B
  h16*   probs    = (h16*)(bb + 37224448);               //  1,048,576 B
  h16*   oh       = (h16*)(bb + 38273024);               //    524,288 B
  h16*   slotsmT  = (h16*)(bb + 38797312);               //    524,288 B
  h16*   slots    = (h16*)(bb + 39321600);               //    524,288 B
  h16*   slotsn   = (h16*)(bb + 39845888);               //    524,288 B
  h16*   vT       = (h16*)(bb + 40370176);               //    524,288 B
  h16*   y2       = (h16*)(bb + 0);
  // BUF_C [4*NCB, 6*NCB): xln | xcat ; later p
  char* bc = w + 4 * NCB;
  h16* xln  = (h16*)(bc + 0);
  h16* xcat = (h16*)(bc + NCB);
  h16* pg   = (h16*)(bc + 0);
  // x1 f32 NCHW
  float* x1 = (float*)(w + 6 * NCB);
  // f16 weights
  char* bw = w + 6 * NCB + (size_t)B_ * C_ * N_ * 4;
  h16* wi_h  = (h16*)(bw + 0);
  h16* wo_h  = (h16*)(bw + 393216);
  h16* l1w_h = (h16*)(bw + 524288);
  h16* l2w_h = (h16*)(bw + 1048576);

  // 0) weights -> f16
  k_f2h<<<(3 * C_ * C_ + 255) / 256, 256, 0, stream>>>(inpw, wi_h, 3 * C_ * C_);
  k_f2h<<<(C_ * C_ + 255) / 256, 256, 0, stream>>>(outpw, wo_h, C_ * C_);
  k_f2h<<<(THID_ * C_ + 255) / 256, 256, 0, stream>>>(lin1w, l1w_h, THID_ * C_);
  k_f2h<<<(C_ * HID_ + 255) / 256, 256, 0, stream>>>(lin2w, l2w_h, C_ * HID_);

  // 1) slot init (pool + l2norm) and LN1 (+ xf / xfn)
  k_pool_slotsn<<<B_ * S_, 256, 0, stream>>>(x, slotsn);
  k_ln1<<<(B_ * N_) / 256, 256, 0, stream>>>(x, n1w, n1b, xf, xfn);

  // 2) logits = scale * slotsn @ xfn^T (then xfn region becomes xfT)
  k_gemm_logits<<<(B_ * (S_ / 16) * (N_ / 64)) / 8, 256, 0, stream>>>(slotsn, xfn, scale, logits);
  k_transp_xf<<<B_ * (N_ / 16) * (C_ / 16), 256, 0, stream>>>(xf, xfT);

  // 3) slots = softmax_n(logits) @ xf
  k_softmax_n<<<B_ * S_, 256, 0, stream>>>(logits, attn);
  k_gemm_slots<<<(B_ * (S_ / 16) * (C_ / 64)) / 8, 256, 0, stream>>>(attn, xfT, slots);

  // 4) MHA on slots
  k_gemm_qkv<<<(B_ * (S_ / 16) * ((3 * C_) / 64)) / 8, 256, 0, stream>>>(slots, wi_h, inpb, qkv);
  k_transp_v<<<(B_ * NH_ * DH_ * S_) / 256, 256, 0, stream>>>(qkv, vT);
  k_gemm_scores<<<(B_ * NH_ * 16) / 8, 256, 0, stream>>>(qkv, scores);
  k_softmax_h<<<(B_ * NH_ * S_ + 255) / 256, 256, 0, stream>>>(scores, probs);
  k_gemm_av<<<(B_ * NH_ * 8) / 8, 256, 0, stream>>>(probs, vT, oh);
  k_gemm_oproj<<<(B_ * (S_ / 16) * (C_ / 64)) / 8, 256, 0, stream>>>(oh, wo_h, outpb, slotsmT);

  // 5) out = softmax_s(logits^T) @ slotsm ; x1 = x + out
  k_softmax_T<<<(B_ * N_) / 256, 256, 0, stream>>>(logits, attnT);
  k_gemm_out<<<(B_ * (N_ / 16) * (C_ / 64)) / 8, 256, 0, stream>>>(attnT, slotsmT, x, x1);

  // 6) FRFN: LN2 -> partial conv -> lin1+gelu -> dwconv*gate -> lin2 (+res)
  k_ln2<<<(B_ * N_) / 256, 256, 0, stream>>>(x1, n2w, n2b, xln);
  k_pconv<<<(B_ * N_ * C_) / 256, 256, 0, stream>>>(xln, pconvw, xcat);
  k_gemm_lin1<<<(B_ * (N_ / 16) * (THID_ / 64)) / 8, 256, 0, stream>>>(xcat, l1w_h, lin1b, y1, y2);
  k_dwconv<<<((size_t)B_ * N_ * HID_) / 256, 256, 0, stream>>>(y1, y2, dww, dwb, pg);
  k_gemm_lin2<<<(B_ * (N_ / 16) * (C_ / 64)) / 8, 256, 0, stream>>>(pg, l2w_h, lin2b, x1, x, out, out2);
}
